// Block_12472585028136
// MI455X (gfx1250) — matrix operand order repaired and checked by reading, not compiled, not run on hardware
//
#include <hip/hip_runtime.h>
#include <hip/hip_bf16.h>

typedef __bf16 bf16_t;
typedef __attribute__((ext_vector_type(16))) __bf16 v16bf;
typedef __attribute__((ext_vector_type(8)))  __bf16 v8bf;
typedef __attribute__((ext_vector_type(8)))  float  v8f;

#define B_    2
#define S_    2048
#define D_    2048
#define QC_   16
#define KVC_  4
#define HD_   128
#define FF_   8192
#define EPS_  1e-5f
#define QKVN_ (D_ + 2 * KVC_ * HD_)   // 3072

__device__ __forceinline__ bf16_t tobf(float x) { return (bf16_t)x; }

__device__ __forceinline__ v8f zero8() {
  v8f z;
#pragma unroll
  for (int i = 0; i < 8; ++i) z[i] = 0.f;
  return z;
}

// Combine two aligned 8xbf16 chunks into one 16xbf16 fragment register set.
__device__ __forceinline__ v16bf pack16(const bf16_t* lo, const bf16_t* hi) {
  v8bf a = *reinterpret_cast<const v8bf*>(lo);
  v8bf b = *reinterpret_cast<const v8bf*>(hi);
  v16bf r;
#pragma unroll
  for (int i = 0; i < 8; ++i) { r[i] = a[i]; r[i + 8] = b[i]; }
  return r;
}

// ---------------------------------------------------------------------------
// Weight prep: W[Kd,Nd] f32 -> Wt[Nd,Kd] bf16 (transposed so WMMA B-fragments
// are contiguous 32B loads).
// ---------------------------------------------------------------------------
__global__ void tcast(const float* __restrict__ W, bf16_t* __restrict__ Wt,
                      int Kd, int Nd) {
  size_t idx = (size_t)blockIdx.x * blockDim.x + threadIdx.x;
  if (idx >= (size_t)Kd * Nd) return;
  int k = (int)(idx / Nd), n = (int)(idx % Nd);
  Wt[(size_t)n * Kd + k] = tobf(W[idx]);
}

// ---------------------------------------------------------------------------
// RMSNorm: one block per row, f32 in -> bf16 out.
// ---------------------------------------------------------------------------
__global__ __launch_bounds__(256)
void rmsnorm_kernel(const float* __restrict__ x, const float* __restrict__ w,
                    bf16_t* __restrict__ out, int Dd) {
  __shared__ float red[256];
  const int row = blockIdx.x;
  const int tid = threadIdx.x;
  const float* xr = x + (size_t)row * Dd;
  float ss = 0.f;
  for (int i = tid; i < Dd; i += 256) { float v = xr[i]; ss += v * v; }
  red[tid] = ss;
  __syncthreads();
  for (int s = 128; s > 0; s >>= 1) {
    if (tid < s) red[tid] += red[tid + s];
    __syncthreads();
  }
  float rr = rsqrtf(red[0] / (float)Dd + EPS_);
  bf16_t* orow = out + (size_t)row * Dd;
  for (int i = tid; i < Dd; i += 256) orow[i] = tobf(xr[i] * rr * w[i]);
}

// ---------------------------------------------------------------------------
// WMMA GEMM: C[M,N] = A[M,K](bf16,row) @ B  with B given transposed
// Bt[N,K](bf16,row).  128 threads = 4 waves; each wave owns a 16x64 C strip.
// MODE 0: C -> f32.  MODE 1: C + Res -> f32.  MODE 2: relu(C)^2 -> bf16.
// ---------------------------------------------------------------------------
template <int MODE>
__global__ __launch_bounds__(128)
void gemm_wmma(const bf16_t* __restrict__ A, const bf16_t* __restrict__ Bt,
               int Kd, int Nd, float* __restrict__ Cf, bf16_t* __restrict__ Cb,
               const float* __restrict__ Res) {
  const int lane = threadIdx.x & 31;
  const int wave = threadIdx.x >> 5;
  const int m0 = blockIdx.y * 16;
  const int n0 = blockIdx.x * 256 + wave * 64;
  const int m  = lane & 15;   // A row within tile / B column lane
  const int kh = lane >> 4;   // half-wave selector in fragment layouts

  v8f acc[4];
#pragma unroll
  for (int t = 0; t < 4; ++t) acc[t] = zero8();

  const bf16_t* arow = A + (size_t)(m0 + m) * Kd;
  for (int k0 = 0; k0 < Kd; k0 += 32) {
    // A fragment 16x32: lane holds row m, k = kh*16 + e (contiguous)
    const bf16_t* ap = arow + k0 + kh * 16;
    v16bf af = pack16(ap, ap + 8);
#pragma unroll
    for (int t = 0; t < 4; ++t) {
      // B fragment 32x16: lane holds col n=m, k = kh*16 + e (contiguous)
      const bf16_t* bp = Bt + (size_t)(n0 + t * 16 + m) * Kd + k0 + kh * 16;
      v16bf bfr = pack16(bp, bp + 8);
      acc[t] = __builtin_amdgcn_wmma_f32_16x16x32_bf16(
          false, af, false, bfr, (short)0, acc[t], false, false);
    }
  }

#pragma unroll
  for (int t = 0; t < 4; ++t) {
#pragma unroll
    for (int r = 0; r < 8; ++r) {
      int row = m0 + 8 * kh + r;
      int col = n0 + t * 16 + m;
      size_t idx = (size_t)row * Nd + col;
      float v = acc[t][r];
      if (MODE == 0) {
        Cf[idx] = v;
      } else if (MODE == 1) {
        Cf[idx] = v + Res[idx];
      } else {
        float rl = v > 0.f ? v : 0.f;
        Cb[idx] = tobf(rl * rl);
      }
    }
  }
}

// ---------------------------------------------------------------------------
// RoPE + repack of the qkv projection:
//   Q[b,h,s,hd]  bf16 (rope applied)
//   K[b,kv,s,hd] bf16 (rope applied, scaled by HD^-0.5)
//   Vt[b,kv,hd,s] bf16 (transposed so attention B-fragments are contiguous)
// ---------------------------------------------------------------------------
__global__ __launch_bounds__(256)
void rope_pack(const float* __restrict__ qkv, const float* __restrict__ sn,
               const float* __restrict__ cs, bf16_t* __restrict__ Q,
               bf16_t* __restrict__ K, bf16_t* __restrict__ V) {
  const int row = blockIdx.x;          // b*S + s
  const int b = row / S_;
  const int s = row % S_;
  const float* src = qkv + (size_t)row * QKVN_;
  const float kscale = 0.08838834764831845f;  // 128^-0.5
  for (int p = threadIdx.x; p < 1536; p += 256) {
    if (p < 1024) {                    // Q: 16 heads * 64 rotation pairs
      int h = p >> 6, i = p & 63;
      float e = src[h * HD_ + 2 * i];
      float o = src[h * HD_ + 2 * i + 1];
      float c = cs[(size_t)s * HD_ + 2 * i];
      float si = sn[(size_t)s * HD_ + 2 * i];
      size_t base = ((size_t)(b * QC_ + h) * S_ + s) * HD_ + 2 * i;
      Q[base]     = tobf(e * c - o * si);
      Q[base + 1] = tobf(o * c + e * si);
    } else if (p < 1280) {             // K: 4 heads * 64 pairs
      int q2 = p - 1024;
      int h = q2 >> 6, i = q2 & 63;
      float e = src[D_ + h * HD_ + 2 * i];
      float o = src[D_ + h * HD_ + 2 * i + 1];
      float c = cs[(size_t)s * HD_ + 2 * i];
      float si = sn[(size_t)s * HD_ + 2 * i];
      size_t base = ((size_t)(b * KVC_ + h) * S_ + s) * HD_ + 2 * i;
      K[base]     = tobf((e * c - o * si) * kscale);
      K[base + 1] = tobf((o * c + e * si) * kscale);
    } else {                           // V: transpose scatter
      int q2 = p - 1280;
      int h = q2 >> 6, i = q2 & 63;
      float v0 = src[D_ + KVC_ * HD_ + h * HD_ + 2 * i];
      float v1 = src[D_ + KVC_ * HD_ + h * HD_ + 2 * i + 1];
      size_t base = ((size_t)(b * KVC_ + h) * HD_ + 2 * i) * S_ + s;
      V[base]      = tobf(v0);
      V[base + S_] = tobf(v1);
    }
  }
}

// ---------------------------------------------------------------------------
// Flash-style causal attention.  Grid: (S/64, QC, B); 128 threads = 4 waves,
// each wave owns 16 query rows and a full 16x128 O accumulator (8 C-frags).
// Scores via WMMA (Q-frags cached in regs, K-frags contiguous), online
// softmax in C-fragment layout (16-lane-half reductions), P staged through
// LDS to become an A-fragment for the P@V WMMAs against Vt.
// ---------------------------------------------------------------------------
__global__ __launch_bounds__(128)
void attn_kernel(const bf16_t* __restrict__ Q, const bf16_t* __restrict__ K,
                 const bf16_t* __restrict__ V, bf16_t* __restrict__ O) {
  __shared__ __align__(16) bf16_t Pl[4][16 * 32];
  const int lane = threadIdx.x & 31;
  const int wave = threadIdx.x >> 5;
  const int m  = lane & 15;
  const int kh = lane >> 4;
  const int q0 = blockIdx.x * 64 + wave * 16;
  const int h  = blockIdx.y;
  const int b  = blockIdx.z;
  const int kvh = h % KVC_;   // reference GQA mapping: head h uses kv head h%KVC

  const bf16_t* Qh = Q + (size_t)(b * QC_ + h) * S_ * HD_;
  const bf16_t* Kh = K + (size_t)(b * KVC_ + kvh) * S_ * HD_;
  const bf16_t* Vh = V + (size_t)(b * KVC_ + kvh) * HD_ * S_;

  v16bf qf[4];
#pragma unroll
  for (int t = 0; t < 4; ++t) {
    const bf16_t* src = Qh + (size_t)(q0 + m) * HD_ + t * 32 + kh * 16;
    qf[t] = pack16(src, src + 8);
  }

  v8f o[8];
#pragma unroll
  for (int nt = 0; nt < 8; ++nt) o[nt] = zero8();
  float mrow[8], lrow[8];
#pragma unroll
  for (int r = 0; r < 8; ++r) { mrow[r] = -1e30f; lrow[r] = 0.f; }

  const int kv_end = blockIdx.x * 64 + 64;  // uniform trip count in block
  for (int kv0 = 0; kv0 < kv_end; kv0 += 32) {
    v8f s0 = zero8(), s1 = zero8();
#pragma unroll
    for (int t = 0; t < 4; ++t) {
      const bf16_t* k0p = Kh + (size_t)(kv0 + m) * HD_ + t * 32 + kh * 16;
      const bf16_t* k1p = Kh + (size_t)(kv0 + 16 + m) * HD_ + t * 32 + kh * 16;
      v16bf b0 = pack16(k0p, k0p + 8);
      v16bf b1 = pack16(k1p, k1p + 8);
      s0 = __builtin_amdgcn_wmma_f32_16x16x32_bf16(
          false, qf[t], false, b0, (short)0, s0, false, false);
      s1 = __builtin_amdgcn_wmma_f32_16x16x32_bf16(
          false, qf[t], false, b1, (short)0, s1, false, false);
    }
    // Online softmax (C layout: lane holds col n=m, rows 8*kh+r)
#pragma unroll
    for (int r = 0; r < 8; ++r) {
      int rowg = q0 + 8 * kh + r;
      float v0 = s0[r] + ((kv0 + m) <= rowg ? 0.f : -1e9f);
      float v1 = s1[r] + ((kv0 + 16 + m) <= rowg ? 0.f : -1e9f);
      float tm = fmaxf(v0, v1);
#pragma unroll
      for (int off = 1; off < 16; off <<= 1)
        tm = fmaxf(tm, __shfl_xor(tm, off, 32));
      float mnew = fmaxf(mrow[r], tm);
      float alpha = __expf(mrow[r] - mnew);
      float p0 = __expf(v0 - mnew);
      float p1 = __expf(v1 - mnew);
      float ts = p0 + p1;
#pragma unroll
      for (int off = 1; off < 16; off <<= 1)
        ts += __shfl_xor(ts, off, 32);
      lrow[r] = lrow[r] * alpha + ts;
      mrow[r] = mnew;
#pragma unroll
      for (int nt = 0; nt < 8; ++nt) o[nt][r] *= alpha;
      Pl[wave][(8 * kh + r) * 32 + m]      = tobf(p0);
      Pl[wave][(8 * kh + r) * 32 + 16 + m] = tobf(p1);
    }
    __syncthreads();
    // Reload P as an A-fragment (16x32 over kv)
    const bf16_t* pls = &Pl[wave][m * 32 + kh * 16];
    v16bf pf = pack16(pls, pls + 8);
#pragma unroll
    for (int nt = 0; nt < 8; ++nt) {
      const bf16_t* vp = Vh + (size_t)(nt * 16 + m) * S_ + kv0 + kh * 16;
      v16bf vb = pack16(vp, vp + 8);
      o[nt] = __builtin_amdgcn_wmma_f32_16x16x32_bf16(
          false, pf, false, vb, (short)0, o[nt], false, false);
    }
    __syncthreads();
  }

  // Write O/l -> attn output laid out [B,S,D] with head h at cols h*HD
#pragma unroll
  for (int nt = 0; nt < 8; ++nt) {
#pragma unroll
    for (int r = 0; r < 8; ++r) {
      int rowg = q0 + 8 * kh + r;
      int col = h * HD_ + nt * 16 + m;
      O[(size_t)(b * S_ + rowg) * D_ + col] = tobf(o[nt][r] / lrow[r]);
    }
  }
}

// ---------------------------------------------------------------------------
// Host-side orchestration (graph-capture safe: only launches on `stream`).
// Inputs: x, mask, rope_sin, rope_cos, wqkv, wo, norm1_w, norm2_w, ffn_wk,
// ffn_wv.  (mask is reproduced analytically: causal -1e9.)
// ---------------------------------------------------------------------------
extern "C" void kernel_launch(void* const* d_in, const int* in_sizes, int n_in,
                              void* d_out, int out_size, void* d_ws,
                              size_t ws_size, hipStream_t stream) {
  (void)in_sizes; (void)n_in; (void)out_size; (void)ws_size;
  const float* x        = (const float*)d_in[0];
  const float* rope_sin = (const float*)d_in[2];
  const float* rope_cos = (const float*)d_in[3];
  const float* wqkv     = (const float*)d_in[4];
  const float* wo       = (const float*)d_in[5];
  const float* n1w      = (const float*)d_in[6];
  const float* n2w      = (const float*)d_in[7];
  const float* wk       = (const float*)d_in[8];
  const float* wv       = (const float*)d_in[9];
  float* out = (float*)d_out;

  char* p = (char*)d_ws;
  auto take = [&](size_t n) {
    char* r = p;
    p += (n + 255) & ~(size_t)255;
    return (void*)r;
  };
  bf16_t* wqkvT = (bf16_t*)take((size_t)QKVN_ * D_ * 2);
  bf16_t* woT   = (bf16_t*)take((size_t)D_ * D_ * 2);
  bf16_t* wkT   = (bf16_t*)take((size_t)FF_ * D_ * 2);
  bf16_t* wvT   = (bf16_t*)take((size_t)D_ * FF_ * 2);
  bf16_t* xn1   = (bf16_t*)take((size_t)B_ * S_ * D_ * 2);
  float*  qkv   = (float*) take((size_t)B_ * S_ * QKVN_ * 4);
  bf16_t* Qb    = (bf16_t*)take((size_t)B_ * QC_ * S_ * HD_ * 2);
  bf16_t* Kb    = (bf16_t*)take((size_t)B_ * KVC_ * S_ * HD_ * 2);
  bf16_t* Vt    = (bf16_t*)take((size_t)B_ * KVC_ * HD_ * S_ * 2);
  bf16_t* ao    = (bf16_t*)take((size_t)B_ * S_ * D_ * 2);
  float*  x1    = (float*) take((size_t)B_ * S_ * D_ * 4);
  bf16_t* xn2   = (bf16_t*)take((size_t)B_ * S_ * D_ * 2);
  bf16_t* hb    = (bf16_t*)take((size_t)B_ * S_ * FF_ * 2);

  const int rows = B_ * S_;  // 4096

  // Weight convert + transpose (bf16, [N,K])
  tcast<<<((size_t)D_ * QKVN_ + 255) / 256, 256, 0, stream>>>(wqkv, wqkvT, D_, QKVN_);
  tcast<<<((size_t)D_ * D_ + 255) / 256, 256, 0, stream>>>(wo, woT, D_, D_);
  tcast<<<((size_t)D_ * FF_ + 255) / 256, 256, 0, stream>>>(wk, wkT, D_, FF_);
  tcast<<<((size_t)FF_ * D_ + 255) / 256, 256, 0, stream>>>(wv, wvT, FF_, D_);

  // x -> rmsnorm -> qkv projection
  rmsnorm_kernel<<<rows, 256, 0, stream>>>(x, n1w, xn1, D_);
  gemm_wmma<0><<<dim3(QKVN_ / 256, rows / 16), 128, 0, stream>>>(
      xn1, wqkvT, D_, QKVN_, qkv, nullptr, nullptr);

  // RoPE + repack
  rope_pack<<<rows, 256, 0, stream>>>(qkv, rope_sin, rope_cos, Qb, Kb, Vt);

  // Causal flash attention
  attn_kernel<<<dim3(S_ / 64, QC_, B_), 128, 0, stream>>>(Qb, Kb, Vt, ao);

  // o @ wo + x  -> x1
  gemm_wmma<1><<<dim3(D_ / 256, rows / 16), 128, 0, stream>>>(
      ao, woT, D_, D_, x1, nullptr, x);

  // x1 -> rmsnorm -> FFN
  rmsnorm_kernel<<<rows, 256, 0, stream>>>(x1, n2w, xn2, D_);
  gemm_wmma<2><<<dim3(FF_ / 256, rows / 16), 128, 0, stream>>>(
      xn2, wkT, D_, FF_, nullptr, hb, nullptr);
  gemm_wmma<1><<<dim3(D_ / 256, rows / 16), 128, 0, stream>>>(
      hb, wvT, FF_, D_, out, nullptr, x1);
}
